// T5Attention_62130996904426
// MI455X (gfx1250) — matrix operand order repaired and checked by reading, not compiled, not run on hardware
//
#include <hip/hip_runtime.h>
#include <hip/hip_bf16.h>

// ---- problem constants (T5 attention) ----
constexpr int Bsz = 2;
constexpr int Sq  = 2048;
constexpr int Dm  = 1024;
constexpr int Hh  = 16;
constexpr int Dkv = 64;

typedef __bf16 bf16_t;
typedef __bf16 v16bf __attribute__((ext_vector_type(16)));
typedef __bf16 v8bf  __attribute__((ext_vector_type(8)));
typedef float  v8f   __attribute__((ext_vector_type(8)));

// wait until this wave's ASYNCcnt <= n (literal)
#define WAIT_ASYNC(n) asm volatile("s_wait_asynccnt " #n ::: "memory")

// per-lane async global->LDS copy of 16 bytes
__device__ inline void async_g2l_b128(unsigned lds_off, unsigned long long gaddr) {
  asm volatile("global_load_async_to_lds_b128 %0, %1, off"
               :: "v"(lds_off), "v"(gaddr) : "memory");
}

__device__ inline bf16_t f2bf(float f) {
  union { float f; unsigned u; } in; in.f = f;
  unsigned u = in.u;
  unsigned r = u + 0x7FFFu + ((u >> 16) & 1u);   // round-to-nearest-even
  union { unsigned short s; bf16_t b; } out;
  out.s = (unsigned short)(r >> 16);
  return out.b;
}

// pack two floats as two bf16 in one u32 (lo = a, hi = b)
__device__ inline unsigned pack_bf2(float a, float b) {
  union { bf16_t h[2]; unsigned u; } o;
  o.h[0] = f2bf(a); o.h[1] = f2bf(b);
  return o.u;
}

__device__ inline v8f wmma_bf(v16bf a, v16bf b, v8f c) {
  return __builtin_amdgcn_wmma_f32_16x16x32_bf16(false, a, false, b, (short)0, c,
                                                 false, false);
}

// A-matrix (16x32 bf16) per-lane fragment from a row-major row pointer.
// elems 0..7  <- K = kc + 16*half + (0..7)
// elems 8..15 <- K = kc + 16*half + 8 + (0..7)
__device__ inline v16bf load_a16(const bf16_t* __restrict__ row, int kc, int half) {
  const bf16_t* p = row + kc + half * 16;
  v8bf lo = *(const v8bf*)(p);
  v8bf hi = *(const v8bf*)(p + 8);
  v16bf r;
#pragma unroll
  for (int i = 0; i < 8; ++i) { r[i] = lo[i]; r[i + 8] = hi[i]; }
  return r;
}

// ------------------------------------------------------------------
// conversion kernels
// ------------------------------------------------------------------
__global__ __launch_bounds__(256) void k_cvt(const float* __restrict__ src,
                                             bf16_t* __restrict__ dst, int n) {
  int i = blockIdx.x * 256 + threadIdx.x;
  if (i < n) dst[i] = f2bf(src[i]);
}

// Wt[n][k] = W[k][n]  (D x D), bf16 -- 32x32 LDS-tiled transpose
__global__ __launch_bounds__(256) void k_cvtT(const float* __restrict__ W,
                                              bf16_t* __restrict__ Wt) {
  __shared__ bf16_t tile[32][33];
  int kt = blockIdx.x >> 5;          // 32x32 tile coords
  int nt = blockIdx.x & 31;
  int c  = threadIdx.x & 31;
  int r8 = threadIdx.x >> 5;         // 0..7
#pragma unroll
  for (int r = 0; r < 4; ++r) {
    int kr = r8 + 8 * r;
    tile[kr][c] = f2bf(W[(size_t)(kt * 32 + kr) * Dm + nt * 32 + c]);
  }
  __syncthreads();
#pragma unroll
  for (int r = 0; r < 4; ++r) {
    int n = r8 + 8 * r;
    Wt[(size_t)(nt * 32 + n) * Dm + kt * 32 + c] = tile[c][n];
  }
}

// ------------------------------------------------------------------
// fused QKV projection: Y = X @ W, scattered to head-major layouts
//   Qh/Kh : [b][h][s][64] bf16     Vt : [b][h][64][s] bf16
// wave task: 16 rows x 64 cols; K looped in 64-chunks, 2x software pipeline
// ------------------------------------------------------------------
__global__ __launch_bounds__(128) void k_qkv(const bf16_t* __restrict__ Xb,
                                             const bf16_t* __restrict__ Wqt,
                                             const bf16_t* __restrict__ Wkt,
                                             const bf16_t* __restrict__ Wvt,
                                             bf16_t* __restrict__ Qh,
                                             bf16_t* __restrict__ Kh,
                                             bf16_t* __restrict__ Vt) {
  const int lane  = threadIdx.x & 31;
  const int wave  = threadIdx.x >> 5;
  const int rlane = lane & 15;
  const int half  = lane >> 4;

  int wid   = blockIdx.x * 4 + wave;          // 0..12287
  int mat   = wid >> 12;                      // 0:q 1:k 2:v
  int rem   = wid & 4095;
  int m0    = (rem >> 4) * 16;                // token tile base (B*S rows)
  int n0    = (rem & 15) * 64;                // output-col tile base

  const bf16_t* Wt = (mat == 0) ? Wqt : (mat == 1) ? Wkt : Wvt;
  const bf16_t* arow = Xb + (size_t)(m0 + rlane) * Dm;
  const bf16_t* brow[4];
#pragma unroll
  for (int t = 0; t < 4; ++t)
    brow[t] = Wt + (size_t)(n0 + t * 16 + rlane) * Dm + half * 16;

  v8f zero = {0.f,0.f,0.f,0.f,0.f,0.f,0.f,0.f};
  v8f acc[4] = {zero, zero, zero, zero};

  // ping/pong in distinct registers: no rotation moves
  v16bf A0 = load_a16(arow, 0, half);
  v16bf B0[4];
#pragma unroll
  for (int t = 0; t < 4; ++t) B0[t] = *(const v16bf*)(brow[t]);

  int kc = 0;
  for (; kc < Dm - 64; kc += 64) {
    v16bf A1 = load_a16(arow, kc + 32, half);
    v16bf B1[4];
#pragma unroll
    for (int t = 0; t < 4; ++t) B1[t] = *(const v16bf*)(brow[t] + kc + 32);
    __builtin_prefetch(arow + kc + 128, 0, 1);
#pragma unroll
    for (int t = 0; t < 4; ++t) acc[t] = wmma_bf(A0, B0[t], acc[t]);
    A0 = load_a16(arow, kc + 64, half);
#pragma unroll
    for (int t = 0; t < 4; ++t) B0[t] = *(const v16bf*)(brow[t] + kc + 64);
#pragma unroll
    for (int t = 0; t < 4; ++t) acc[t] = wmma_bf(A1, B1[t], acc[t]);
  }
  {
    v16bf A1 = load_a16(arow, kc + 32, half);
    v16bf B1[4];
#pragma unroll
    for (int t = 0; t < 4; ++t) B1[t] = *(const v16bf*)(brow[t] + kc + 32);
#pragma unroll
    for (int t = 0; t < 4; ++t) acc[t] = wmma_bf(A0, B0[t], acc[t]);
#pragma unroll
    for (int t = 0; t < 4; ++t) acc[t] = wmma_bf(A1, B1[t], acc[t]);
  }

#pragma unroll
  for (int t = 0; t < 4; ++t) {
#pragma unroll
    for (int i = 0; i < 8; ++i) {
      int token = m0 + i + 8 * half;
      int b = token >> 11, s = token & (Sq - 1);
      int n = n0 + t * 16 + rlane;
      int h = n >> 6, dd = n & 63;
      bf16_t v = f2bf(acc[t][i]);
      size_t bh = (size_t)b * Hh + h;
      if (mat == 0)      Qh[(bh * Sq + s) * Dkv + dd] = v;
      else if (mat == 1) Kh[(bh * Sq + s) * Dkv + dd] = v;
      else               Vt[(bh * Dkv + dd) * Sq + s] = v;
    }
  }
}

// ------------------------------------------------------------------
// attention: block = 4 waves sharing one (b,h); each wave owns 16 q-rows.
// K/V tiles staged into LDS with double-buffered async copies.
// ------------------------------------------------------------------
__global__ __launch_bounds__(128) void k_attn(const bf16_t* __restrict__ Qh,
                                              const bf16_t* __restrict__ Kh,
                                              const bf16_t* __restrict__ Vt,
                                              const int*    __restrict__ mask,
                                              const float*  __restrict__ rel_bias,
                                              float*        __restrict__ attn,
                                              bf16_t*       __restrict__ Ctx) {
  __shared__ __align__(32) bf16_t kbuf[2][32][64];   // 8 KB ping-pong, rows k, cols d
  __shared__ __align__(32) bf16_t vbuf[2][64][32];   // 8 KB ping-pong, rows d, cols k
  __shared__ float biasT[32];
  __shared__ __align__(16) unsigned pk[4][16][20];   // per-wave packed-P staging

  const int tid   = threadIdx.x;
  const int lane  = tid & 31;
  const int wave  = tid >> 5;
  const int rlane = lane & 15;
  const int half  = lane >> 4;

  const int qt = blockIdx.x & 31;              // S/64 = 32 q-groups
  const int bh = blockIdx.x >> 5;
  const int b  = bh >> 4;
  const int h  = bh & 15;

  if (tid < 32) biasT[tid] = rel_bias[tid * Hh + h];
  __syncthreads();

  const int q0 = qt * 64 + wave * 16;
  const bf16_t* Qbase = Qh + ((size_t)bh * Sq + q0) * Dkv;
  const bf16_t* Kbase = Kh + (size_t)bh * Sq * Dkv;
  const bf16_t* Vbase = Vt + (size_t)bh * Dkv * Sq;
  const int*    mrow  = mask + b * Sq;

  // per-lane mask bitmask: bit i -> mask[i*16 + rlane]  (lane only sees k%16==rlane)
  unsigned mb[4] = {0u, 0u, 0u, 0u};
  for (int i = 0; i < 128; ++i)
    mb[i >> 5] |= (mrow[i * 16 + rlane] != 0 ? 1u : 0u) << (i & 31);

  const unsigned kbufL = (unsigned)(size_t)&kbuf[0][0][0];
  const unsigned vbufL = (unsigned)(size_t)&vbuf[0][0][0];

  // cooperative async copy of K tile (rows k0..k0+31, 4KB contiguous): 2 instr
  auto copyK = [&](int k0, int buf) {
    unsigned long long g = (unsigned long long)(size_t)(Kbase + (size_t)k0 * Dkv);
#pragma unroll
    for (int c = 0; c < 2; ++c)
      async_g2l_b128(kbufL + buf * 4096 + tid * 16 + c * 2048,
                     g + tid * 16 + c * 2048);
  };
  // cooperative async copy of V tile (rows d=0..63, cols k0..k0+31): 2 instr
  auto copyV = [&](int k0, int buf) {
    unsigned long long g = (unsigned long long)(size_t)(Vbase + k0);
#pragma unroll
    for (int c = 0; c < 2; ++c) {
      int id  = tid + 128 * c;          // 0..255 -> 16B chunks
      int row = id >> 2, seg = id & 3;
      async_g2l_b128(vbufL + buf * 4096 + row * 64 + seg * 16,
                     g + (size_t)row * Sq * 2 + seg * 16);
    }
  };

  // Q fragments (A layout), two 32-wide chunks covering d=0..63
  v16bf Aq0 = load_a16(Qbase + rlane * Dkv, 0,  half);
  v16bf Aq1 = load_a16(Qbase + rlane * Dkv, 32, half);

  v8f zero = {0.f,0.f,0.f,0.f,0.f,0.f,0.f,0.f};

  // score tile from LDS K buffer; j = 32-wide tile index, sub in {0,1}
  auto score16 = [&](int buf, int sub, int j) -> v8f {
    const bf16_t* kp = &kbuf[buf][sub * 16 + rlane][0];
    v8f acc = zero;
    acc = wmma_bf(Aq0, *(const v16bf*)(kp + half * 16), acc);
    acc = wmma_bf(Aq1, *(const v16bf*)(kp + 32 + half * 16), acc);
    int idx = 2 * j + sub;                   // k = idx*16 + rlane
    int k = idx * 16 + rlane;
    bool ok = ((mb[idx >> 5] >> (idx & 31)) & 1u) != 0u;
    v8f s;
#pragma unroll
    for (int i = 0; i < 8; ++i) {
      int q = q0 + i + 8 * half;
      int d = q - k;                                    // n = -(k - q)
      int ret = (d < 0) ? 16 : 0;
      int a = (d < 0) ? -d : d;
      int bk;
      if (a < 8) bk = ret + a;
      else {
        int v = 8 + (int)(2.0f * __log2f((float)a * 0.125f));
        bk = ret + (v > 15 ? 15 : v);
      }
      s[i] = ok ? (acc[i] * 0.125f + biasT[bk]) : -1e9f;
    }
    return s;
  };

  constexpr int NT = Sq / 32;   // 64 tiles of 32 k

  // ================= pass 1: row max (K only staged) =================
  float rmax[8];
#pragma unroll
  for (int i = 0; i < 8; ++i) rmax[i] = -1e30f;

  copyK(0, 0);
  copyK(32, 1);
  for (int j = 0; j < NT; ++j) {
    if (j + 1 < NT) { WAIT_ASYNC(2); } else { WAIT_ASYNC(0); }
    __syncthreads();
    int buf = j & 1;
    v8f s0 = score16(buf, 0, j);
    v8f s1 = score16(buf, 1, j);
#pragma unroll
    for (int i = 0; i < 8; ++i)
      rmax[i] = fmaxf(rmax[i], fmaxf(s0[i], s1[i]));
    __syncthreads();
    if (j + 2 < NT) copyK((j + 2) * 32, buf);
  }
#pragma unroll
  for (int off = 1; off < 16; off <<= 1)
#pragma unroll
    for (int i = 0; i < 8; ++i)
      rmax[i] = fmaxf(rmax[i], __shfl_xor(rmax[i], off, 32));

  // ===== pass 2: e = exp(s - M), row sums, unnormalized PV (K+V staged) =====
  float rsum[8] = {0.f,0.f,0.f,0.f,0.f,0.f,0.f,0.f};
  v8f cacc[4] = {zero, zero, zero, zero};
  unsigned (*pr)[20] = pk[wave];

  copyK(0, 0);  copyV(0, 0);
  copyK(32, 1); copyV(32, 1);
  for (int j = 0; j < NT; ++j) {
    if (j + 1 < NT) { WAIT_ASYNC(4); } else { WAIT_ASYNC(0); }
    __syncthreads();
    int buf = j & 1;
    v8f e0 = score16(buf, 0, j);
    v8f e1 = score16(buf, 1, j);
#pragma unroll
    for (int i = 0; i < 8; ++i) {
      e0[i] = __expf(e0[i] - rmax[i]);
      e1[i] = __expf(e1[i] - rmax[i]);
      rsum[i] += e0[i] + e1[i];
    }
    // C-layout -> LDS (packed bf16 pairs) -> A-layout, wave-synchronous.
    // u32 at pr[m][k'] holds (e at k=k', e at k=k'+16): exactly A elems (t, t+8).
#pragma unroll
    for (int i = 0; i < 8; ++i)
      pr[i + 8 * half][rlane] = pack_bf2(e0[i], e1[i]);
    __builtin_amdgcn_wave_barrier();
    const uint4* lp = (const uint4*)&pr[rlane][half * 8];
    uint4 w0 = lp[0], w1 = lp[1];
    __builtin_amdgcn_wave_barrier();
    v16bf Ae;
    {
      unsigned w[8] = {w0.x, w0.y, w0.z, w0.w, w1.x, w1.y, w1.z, w1.w};
#pragma unroll
      for (int t = 0; t < 8; ++t) {
        union { unsigned u; bf16_t h[2]; } x; x.u = w[t];
        Ae[t]     = x.h[0];
        Ae[t + 8] = x.h[1];
      }
    }
#pragma unroll
    for (int t = 0; t < 4; ++t) {
      const bf16_t* vp = &vbuf[buf][t * 16 + rlane][half * 8];
      v8bf vlo = *(const v8bf*)vp, vhi = *(const v8bf*)(vp + 16);
      v16bf Bv;
#pragma unroll
      for (int i = 0; i < 8; ++i) { Bv[i] = vlo[i]; Bv[i + 8] = vhi[i]; }
      cacc[t] = wmma_bf(Ae, Bv, cacc[t]);
    }
    __syncthreads();
    if (j + 2 < NT) { copyK((j + 2) * 32, buf); copyV((j + 2) * 32, buf); }
  }
#pragma unroll
  for (int off = 1; off < 16; off <<= 1)
#pragma unroll
    for (int i = 0; i < 8; ++i) rsum[i] += __shfl_xor(rsum[i], off, 32);
  float rinv[8];
#pragma unroll
  for (int i = 0; i < 8; ++i) rinv[i] = 1.0f / rsum[i];

  // ========== pass 3: write normalized probabilities (K staged) ==========
  float* arow = attn + ((size_t)bh * Sq + q0) * Sq;
  copyK(0, 0);
  copyK(32, 1);
  for (int j = 0; j < NT; ++j) {
    if (j + 1 < NT) { WAIT_ASYNC(2); } else { WAIT_ASYNC(0); }
    __syncthreads();
    int buf = j & 1;
    int k0 = j * 32;
    v8f p0 = score16(buf, 0, j);
    v8f p1 = score16(buf, 1, j);
#pragma unroll
    for (int i = 0; i < 8; ++i) {
      float* r = arow + (size_t)(i + 8 * half) * Sq + k0 + rlane;
      r[0]  = __expf(p0[i] - rmax[i]) * rinv[i];
      r[16] = __expf(p1[i] - rmax[i]) * rinv[i];
    }
    __syncthreads();
    if (j + 2 < NT) copyK((j + 2) * 32, buf);
  }

  // ---- store context (normalized) as bf16 [b][s][h*64+dd] ----
#pragma unroll
  for (int t = 0; t < 4; ++t) {
#pragma unroll
    for (int i = 0; i < 8; ++i) {
      int s = q0 + i + 8 * half;
      int col = h * Dkv + t * 16 + rlane;
      Ctx[((size_t)b * Sq + s) * Dm + col] = f2bf(cacc[t][i] * rinv[i]);
    }
  }
}

// ------------------------------------------------------------------
// output projection: out = Ctx @ Wo  (fp32 result), 2x software pipeline
// ------------------------------------------------------------------
__global__ __launch_bounds__(128) void k_out(const bf16_t* __restrict__ Ctx,
                                             const bf16_t* __restrict__ Wot,
                                             float* __restrict__ outp) {
  const int lane  = threadIdx.x & 31;
  const int wave  = threadIdx.x >> 5;
  const int rlane = lane & 15;
  const int half  = lane >> 4;

  int wid = blockIdx.x * 4 + wave;           // 0..4095
  int m0  = (wid >> 4) * 16;
  int n0  = (wid & 15) * 64;

  const bf16_t* arow = Ctx + (size_t)(m0 + rlane) * Dm;
  const bf16_t* brow[4];
#pragma unroll
  for (int t = 0; t < 4; ++t)
    brow[t] = Wot + (size_t)(n0 + t * 16 + rlane) * Dm + half * 16;

  v8f zero = {0.f,0.f,0.f,0.f,0.f,0.f,0.f,0.f};
  v8f acc[4] = {zero, zero, zero, zero};

  v16bf A0 = load_a16(arow, 0, half);
  v16bf B0[4];
#pragma unroll
  for (int t = 0; t < 4; ++t) B0[t] = *(const v16bf*)(brow[t]);

  int kc = 0;
  for (; kc < Dm - 64; kc += 64) {
    v16bf A1 = load_a16(arow, kc + 32, half);
    v16bf B1[4];
#pragma unroll
    for (int t = 0; t < 4; ++t) B1[t] = *(const v16bf*)(brow[t] + kc + 32);
    __builtin_prefetch(arow + kc + 128, 0, 1);
#pragma unroll
    for (int t = 0; t < 4; ++t) acc[t] = wmma_bf(A0, B0[t], acc[t]);
    A0 = load_a16(arow, kc + 64, half);
#pragma unroll
    for (int t = 0; t < 4; ++t) B0[t] = *(const v16bf*)(brow[t] + kc + 64);
#pragma unroll
    for (int t = 0; t < 4; ++t) acc[t] = wmma_bf(A1, B1[t], acc[t]);
  }
  {
    v16bf A1 = load_a16(arow, kc + 32, half);
    v16bf B1[4];
#pragma unroll
    for (int t = 0; t < 4; ++t) B1[t] = *(const v16bf*)(brow[t] + kc + 32);
#pragma unroll
    for (int t = 0; t < 4; ++t) acc[t] = wmma_bf(A0, B0[t], acc[t]);
#pragma unroll
    for (int t = 0; t < 4; ++t) acc[t] = wmma_bf(A1, B1[t], acc[t]);
  }

#pragma unroll
  for (int t = 0; t < 4; ++t) {
#pragma unroll
    for (int i = 0; i < 8; ++i) {
      int token = m0 + i + 8 * half;
      int n = n0 + t * 16 + rlane;
      outp[(size_t)token * Dm + n] = acc[t][i];
    }
  }
}

// ------------------------------------------------------------------
extern "C" void kernel_launch(void* const* d_in, const int* in_sizes, int n_in,
                              void* d_out, int out_size, void* d_ws, size_t ws_size,
                              hipStream_t stream) {
  const float* hidden   = (const float*)d_in[0];
  const int*   mask     = (const int*)  d_in[1];
  const float* Wq       = (const float*)d_in[2];
  const float* Wk       = (const float*)d_in[3];
  const float* Wv       = (const float*)d_in[4];
  const float* Wo       = (const float*)d_in[5];
  const float* rel_bias = (const float*)d_in[6];

  float* outp = (float*)d_out;                               // [B,S,D]
  float* attn = outp + (size_t)Bsz * Sq * Dm;                // [B,H,S,S]

  char* ws = (char*)d_ws;
  const size_t nBSD = (size_t)Bsz * Sq * Dm;                 // 4,194,304
  bf16_t* Xb  = (bf16_t*)(ws);                               // 8 MB
  bf16_t* Wqt = (bf16_t*)(ws + 8  * 1024 * 1024);
  bf16_t* Wkt = (bf16_t*)(ws + 10 * 1024 * 1024);
  bf16_t* Wvt = (bf16_t*)(ws + 12 * 1024 * 1024);
  bf16_t* Wot = (bf16_t*)(ws + 14 * 1024 * 1024);
  bf16_t* Qh  = (bf16_t*)(ws + 16 * 1024 * 1024);            // 8 MB
  bf16_t* Kh  = (bf16_t*)(ws + 24 * 1024 * 1024);            // 8 MB
  bf16_t* Vt  = (bf16_t*)(ws + 32 * 1024 * 1024);            // 8 MB
  bf16_t* Ctx = (bf16_t*)(ws + 40 * 1024 * 1024);            // 8 MB

  k_cvt <<<(int)(nBSD / 256), 256, 0, stream>>>(hidden, Xb, (int)nBSD);
  k_cvtT<<<1024, 256, 0, stream>>>(Wq, Wqt);
  k_cvtT<<<1024, 256, 0, stream>>>(Wk, Wkt);
  k_cvtT<<<1024, 256, 0, stream>>>(Wv, Wvt);
  k_cvtT<<<1024, 256, 0, stream>>>(Wo, Wot);

  k_qkv <<<3072, 128, 0, stream>>>(Xb, Wqt, Wkt, Wvt, Qh, Kh, Vt);
  k_attn<<<1024, 128, 0, stream>>>(Qh, Kh, Vt, mask, rel_bias, attn, Ctx);
  k_out <<<1024, 128, 0, stream>>>(Ctx, Wot, outp);
}
